// Net_87926570484115
// MI455X (gfx1250) — matrix operand order repaired and checked by reading, not compiled, not run on hardware
//
#include <hip/hip_runtime.h>
#include <hip/hip_bf16.h>
#include <math.h>

// Problem constants (match reference)
constexpr int kSS = 17;          // single_state
constexpr int kNA = 12;          // n_action
constexpr int kNB = 3;           // n_blocks
constexpr int kVD = 64;          // vector_dim
constexpr int kXK = 32;          // padded feature K for QKV (17+12 -> 32)
constexpr float kEps = 1e-8f;

// Tiling
constexpr int kWaves = 2;                      // waves per block (wave32)
constexpr int kItemsPerWave = 16;              // batch items per wave
constexpr int kRows = kItemsPerWave * kNB;     // 48 rows per wave (3 M-tiles)
constexpr int kThreads = kWaves * 32;

typedef __attribute__((ext_vector_type(16))) _Float16 v16h;
typedef __attribute__((ext_vector_type(8)))  _Float16 v8h;
typedef __attribute__((ext_vector_type(8)))  float    v8f;

__device__ __forceinline__ v8f wmma_f16(v16h a, v16h b, v8f c) {
  // D(16x16,f32) = A(16x32,f16) * B(32x16,f16) + C
  return __builtin_amdgcn_wmma_f32_16x16x32_f16(false, a, false, b, (short)0, c,
                                                false, false);
}

// Load a pre-swizzled B fragment (16 contiguous halves per lane).
__device__ __forceinline__ v16h load_frag(const _Float16* p) {
  v16h r;
  ((uint4*)&r)[0] = ((const uint4*)p)[0];
  ((uint4*)&r)[1] = ((const uint4*)p)[1];
  return r;
}

// Build a 16x32 f16 A fragment from a row-major f16 LDS buffer.
// Contiguous k order used here: lane L holds M = L&15; halves 0-15 -> K = kb..kb+15,
// kb = 16*(L>>4).
__device__ __forceinline__ v16h load_fragA(const _Float16* buf, int rowBase,
                                           int stride, int k0, int lane) {
  const int m  = lane & 15;
  const int kb = (lane >> 4) << 4;
  const _Float16* p = buf + (rowBase + m) * stride + k0 + kb;
  v16h r;
  ((uint4*)&r)[0] = *(const uint4*)(p);
  ((uint4*)&r)[1] = *(const uint4*)(p + 8);
  return r;
}

__device__ __forceinline__ float dot8(v8h a, v8h b) {
  float s = 0.f;
#pragma unroll
  for (int e = 0; e < 8; ++e) s += (float)a[e] * (float)b[e];
  return s;
}

__global__ __launch_bounds__(kThreads, 1) void fused_block_attn_kernel(
    const float* __restrict__ states, const int* __restrict__ block_id,
    const float* __restrict__ action,
    const float* __restrict__ Wq, const float* __restrict__ bq,
    const float* __restrict__ Wk, const float* __restrict__ bk,
    const float* __restrict__ Wv, const float* __restrict__ bv,
    const float* __restrict__ W1, const float* __restrict__ b1,
    const float* __restrict__ W2, const float* __restrict__ b2,
    float* __restrict__ out, int Btot) {
  // Pre-swizzled B fragments (512 halves per 16x16 N-tile / 32-K-slice)
  __shared__ __align__(16) _Float16 s_fragQ[4 * 512];
  __shared__ __align__(16) _Float16 s_fragK[4 * 512];
  __shared__ __align__(16) _Float16 s_fragV[4 * 512];
  __shared__ __align__(16) _Float16 s_frag1[8 * 512];   // W1: 4 nt x 2 kt
  __shared__ __align__(16) _Float16 s_frag2[4 * 512];   // W2: 2 nt x 2 kt
  // Per-wave staging (f16). bufQ doubles as X staging and residual staging;
  // bufK doubles as the tanh-hidden staging.
  __shared__ __align__(16) _Float16 s_bufQ[kWaves][kRows * kVD];
  __shared__ __align__(16) _Float16 s_bufK[kWaves][kRows * kVD];
  __shared__ __align__(16) _Float16 s_bufV[kWaves][kRows * kVD];

  const int tid  = threadIdx.x;
  const int wave = tid >> 5;
  const int lane = tid & 31;

  // ---- Stage QKV weights as B fragments: n = 16*nt + (L&15), k = h + 16*(L>>4)
  for (int idx = tid; idx < 4 * 512; idx += kThreads) {
    const int f = idx >> 9, L = (idx >> 4) & 31, h = idx & 15;
    const int n = (f << 4) | (L & 15);
    const int k = h + ((L >> 4) << 4);
    const bool ok = (k < kSS + kNA);
    s_fragQ[idx] = (_Float16)(ok ? Wq[n * (kSS + kNA) + k] : 0.f);
    s_fragK[idx] = (_Float16)(ok ? Wk[n * (kSS + kNA) + k] : 0.f);
    s_fragV[idx] = (_Float16)(ok ? Wv[n * (kSS + kNA) + k] : 0.f);
  }
  for (int idx = tid; idx < 8 * 512; idx += kThreads) {
    const int f = idx >> 9, L = (idx >> 4) & 31, h = idx & 15;
    const int nt = f >> 1, kt = f & 1;
    const int n = (nt << 4) | (L & 15);
    const int k = (kt << 5) + h + ((L >> 4) << 4);
    s_frag1[idx] = (_Float16)W1[n * kVD + k];
  }
  for (int idx = tid; idx < 4 * 512; idx += kThreads) {
    const int f = idx >> 9, L = (idx >> 4) & 31, h = idx & 15;
    const int nt = f >> 1, kt = f & 1;
    const int n = (nt << 4) | (L & 15);
    const int k = (kt << 5) + h + ((L >> 4) << 4);
    s_frag2[idx] = (_Float16)((n < kSS) ? W2[n * kVD + k] : 0.f);
  }

  const int itemBase = blockIdx.x * (kWaves * kItemsPerWave) + wave * kItemsPerWave;
  const int rowBase  = itemBase * kNB;

  // ---- Stage X = concat(states, action_dummy) for this wave's 48 rows (K padded to 32)
  if (itemBase < Btot) {
    __builtin_prefetch(states + (size_t)rowBase * kSS, 0, 1);
    for (int idx = lane; idx < kRows * kXK; idx += 32) {
      const int r = idx >> 5, k = idx & 31;
      const int grow = rowBase + r;
      float val;
      if (k < kSS) {
        val = states[grow * kSS + k];
      } else if (k < kSS + kNA) {
        const int b = itemBase + r / 3;
        val = (block_id[grow] == 1) ? action[b * kNA + (k - kSS)] : -1.0f;
      } else {
        val = 0.f;
      }
      s_bufQ[wave][idx] = (_Float16)val;   // X overlays bufQ (stride 32 halves)
    }
  }
  __syncthreads();
  if (itemBase >= Btot) return;

  const int ncolLo = lane & 15;
  const int mOfs   = (lane >> 4) << 3;

  // ---- QKV projections: preload A-frags (X is about to be overwritten by Q)
  v16h aX[3];
#pragma unroll
  for (int t = 0; t < 3; ++t)
    aX[t] = load_fragA(&s_bufQ[wave][0], t * 16, kXK, 0, lane);

#pragma unroll
  for (int t = 0; t < 3; ++t) {
#pragma unroll
    for (int nt = 0; nt < 4; ++nt) {
      v8f cq = {}, ck = {}, cv = {};
      cq = wmma_f16(aX[t], load_frag(&s_fragQ[nt * 512 + lane * 16]), cq);
      ck = wmma_f16(aX[t], load_frag(&s_fragK[nt * 512 + lane * 16]), ck);
      cv = wmma_f16(aX[t], load_frag(&s_fragV[nt * 512 + lane * 16]), cv);
      const int ncol = (nt << 4) | ncolLo;
      const float bqv = bq[ncol], bkv = bk[ncol], bvv = bv[ncol];
#pragma unroll
      for (int r = 0; r < 8; ++r) {
        const int row = t * 16 + r + mOfs;     // C layout: M = r + 8*(L>>4)
        s_bufQ[wave][row * kVD + ncol] = (_Float16)(cq[r] + bqv);
        s_bufK[wave][row * kVD + ncol] = (_Float16)(ck[r] + bkv);
        s_bufV[wave][row * kVD + ncol] = (_Float16)(cv[r] + bvv);
      }
    }
  }

  // ---- Cosine-similarity attention: 2 lanes per item, each lane half of VD
  {
    const int it = lane >> 1;
    const int ph = lane & 1;
    const int r0 = it * 3;
    const _Float16* Qb = &s_bufQ[wave][0];
    const _Float16* Kb = &s_bufK[wave][0];
    const _Float16* Vb = &s_bufV[wave][0];
    float num[3][3] = {}, qn[3] = {}, kn[3] = {};
#pragma unroll
    for (int n = 0; n < 3; ++n) {
      v8h qc[4];
#pragma unroll
      for (int c = 0; c < 4; ++c) {
        qc[c] = ((const v8h*)&Qb[(r0 + n) * kVD])[ph * 4 + c];
        qn[n] += dot8(qc[c], qc[c]);
      }
#pragma unroll
      for (int j = 0; j < 3; ++j) {
#pragma unroll
        for (int c = 0; c < 4; ++c) {
          const v8h kc = ((const v8h*)&Kb[(r0 + j) * kVD])[ph * 4 + c];
          num[n][j] += dot8(qc[c], kc);
          if (n == 0) kn[j] += dot8(kc, kc);
        }
      }
    }
#pragma unroll
    for (int n = 0; n < 3; ++n) {
      qn[n] += __shfl_xor(qn[n], 1, 32);
      kn[n] += __shfl_xor(kn[n], 1, 32);
#pragma unroll
      for (int j = 0; j < 3; ++j) num[n][j] += __shfl_xor(num[n][j], 1, 32);
    }
#pragma unroll
    for (int n = 0; n < 3; ++n) {
      float d[3];
#pragma unroll
      for (int j = 0; j < 3; ++j)
        d[j] = num[n][j] / fmaxf(sqrtf(qn[n]) * sqrtf(kn[j]), kEps);
      const float mx = fmaxf(d[0], fmaxf(d[1], d[2]));
      const float e0 = expf(d[0] - mx), e1 = expf(d[1] - mx), e2 = expf(d[2] - mx);
      const float inv = 1.f / (e0 + e1 + e2);
      const float a[3] = {e0 * inv, e1 * inv, e2 * inv};
#pragma unroll
      for (int c = 0; c < 4; ++c) {
        float acc[8] = {};
#pragma unroll
        for (int j = 0; j < 3; ++j) {
          const v8h vc = ((const v8h*)&Vb[(r0 + j) * kVD])[ph * 4 + c];
#pragma unroll
          for (int e = 0; e < 8; ++e) acc[e] += a[j] * (float)vc[e];
        }
        v8h o;
#pragma unroll
        for (int e = 0; e < 8; ++e) o[e] = (_Float16)acc[e];
        // residual overlays bufQ (all Q reads above precede these writes)
        ((v8h*)&s_bufQ[wave][(r0 + n) * kVD])[ph * 4 + c] = o;
      }
    }
  }

  // ---- W1 (64x64) + tanh: residual(bufQ) -> hidden(bufK)
#pragma unroll
  for (int t = 0; t < 3; ++t) {
    const v16h a0 = load_fragA(&s_bufQ[wave][0], t * 16, kVD, 0, lane);
    const v16h a1 = load_fragA(&s_bufQ[wave][0], t * 16, kVD, 32, lane);
#pragma unroll
    for (int nt = 0; nt < 4; ++nt) {
      v8f c = {};
      c = wmma_f16(a0, load_frag(&s_frag1[(nt * 2 + 0) * 512 + lane * 16]), c);
      c = wmma_f16(a1, load_frag(&s_frag1[(nt * 2 + 1) * 512 + lane * 16]), c);
      const int ncol = (nt << 4) | ncolLo;
      const float bias = b1[ncol];
#pragma unroll
      for (int r = 0; r < 8; ++r) {
        const int row = t * 16 + r + mOfs;
        s_bufK[wave][row * kVD + ncol] = (_Float16)tanhf(c[r] + bias);
      }
    }
  }

  // ---- W2 (17x64) + bias + states residual -> out (flat index row*17 + col)
#pragma unroll
  for (int t = 0; t < 3; ++t) {
    const v16h a0 = load_fragA(&s_bufK[wave][0], t * 16, kVD, 0, lane);
    const v16h a1 = load_fragA(&s_bufK[wave][0], t * 16, kVD, 32, lane);
#pragma unroll
    for (int nt = 0; nt < 2; ++nt) {
      v8f c = {};
      c = wmma_f16(a0, load_frag(&s_frag2[(nt * 2 + 0) * 512 + lane * 16]), c);
      c = wmma_f16(a1, load_frag(&s_frag2[(nt * 2 + 1) * 512 + lane * 16]), c);
      const int col = (nt << 4) | ncolLo;
      if (col < kSS) {
        const float bias = b2[col];
#pragma unroll
        for (int r = 0; r < 8; ++r) {
          const int row = rowBase + t * 16 + r + mOfs;
          out[row * kSS + col] = c[r] + bias + states[row * kSS + col];
        }
      }
    }
  }
}

extern "C" void kernel_launch(void* const* d_in, const int* in_sizes, int n_in,
                              void* d_out, int out_size, void* d_ws, size_t ws_size,
                              hipStream_t stream) {
  const float* states   = (const float*)d_in[0];
  const int*   block_id = (const int*)d_in[1];
  const float* action   = (const float*)d_in[2];
  const float* Wq = (const float*)d_in[3];
  const float* bq = (const float*)d_in[4];
  const float* Wk = (const float*)d_in[5];
  const float* bk = (const float*)d_in[6];
  const float* Wv = (const float*)d_in[7];
  const float* bv = (const float*)d_in[8];
  const float* W1 = (const float*)d_in[9];
  const float* b1 = (const float*)d_in[10];
  const float* W2 = (const float*)d_in[11];
  const float* b2 = (const float*)d_in[12];
  float* out = (float*)d_out;

  const int Btot = in_sizes[0] / (kNB * kSS);
  const int itemsPerBlock = kWaves * kItemsPerWave;       // 32 items / block
  const int grid = (Btot + itemsPerBlock - 1) / itemsPerBlock;

  hipLaunchKernelGGL(fused_block_attn_kernel, dim3(grid), dim3(kThreads), 0,
                     stream, states, block_id, action, Wq, bq, Wk, bk, Wv, bv,
                     W1, b1, W2, b2, out, Btot);
}
